// FineGrainedCellLSTM_80324478370227
// MI455X (gfx1250) — matrix operand order repaired and checked by reading, not compiled, not run on hardware
//
#include <hip/hip_runtime.h>
#include <hip/hip_bf16.h>

#define T_ 512
#define B_ 64
#define I_ 1024
#define H_ 1024
#define G_ 4096   // 4*H

typedef __bf16 bf16;
typedef __attribute__((ext_vector_type(16))) __bf16 bf16x16;
typedef __attribute__((ext_vector_type(8)))  __bf16 bf16x8;
typedef __attribute__((ext_vector_type(8)))  float  f32x8;

#if __has_builtin(__builtin_amdgcn_global_load_async_to_lds_b128) && \
    __has_builtin(__builtin_amdgcn_s_wait_asynccnt)
#define USE_ASYNC_LDS 1
#else
#define USE_ASYNC_LDS 0
#endif

typedef int v4i_t __attribute__((vector_size(16)));
typedef __attribute__((address_space(1))) v4i_t* gv4i_p;   // global (AS1)
typedef __attribute__((address_space(3))) v4i_t* lv4i_p;   // LDS (AS3)

// 16-byte global -> LDS copy; async (ASYNCcnt-tracked) when available.
__device__ __forceinline__ void copy16_g2l(float* lds_dst, const float* gsrc) {
#if USE_ASYNC_LDS
    __builtin_amdgcn_global_load_async_to_lds_b128(
        (gv4i_p)gsrc, (lv4i_p)lds_dst, 0, 0);
#else
    *(float4*)lds_dst = *(const float4*)gsrc;
#endif
}

__device__ __forceinline__ void wait_g2l() {
#if USE_ASYNC_LDS
    __builtin_amdgcn_s_wait_asynccnt(0);
#endif
}

union AFrag { bf16x16 v; struct { bf16x8 lo; bf16x8 hi; } h; };

// A-matrix 16x32 bf16 fragment from row-major memory.
// ISA layout: lanes 0-15 row M=lane, V0..V3 = K 0..7, V4..V7 = K 16..23;
// lanes 16-31 same rows, K 8..15 and K 24..31.
__device__ __forceinline__ bf16x16 load_a_frag(const bf16* __restrict__ A, int lda,
                                               int mbase, int kbase, int lane) {
    const int row = mbase + (lane & 15);
    const int k0  = kbase + ((lane >> 4) << 3);     // +0 or +8
    const bf16* p = A + (size_t)row * lda + k0;
    AFrag f;
    f.h.lo = *(const bf16x8*)(p);                   // K = k0 .. k0+7
    f.h.hi = *(const bf16x8*)(p + 16);              // K = k0+16 .. k0+23
    return f.v;
}

// B-matrix 32x16 bf16 fragment, pre-packed so each lane's 16 values are contiguous.
__device__ __forceinline__ bf16x16 load_b_frag(const bf16* __restrict__ P, int ntile,
                                               int kt, int nkt, int lane) {
    const bf16* p = P + ((size_t)(ntile * nkt + kt) << 9) + (lane << 4);
    return *(const bf16x16*)p;   // 32 B per lane -> 2x global_load_b128
}

// ---- Weight packing: W (K x N, row-major f32) -> bf16 WMMA B-fragments ----
// frag(kt,nt): lane l holds column nt*16+(l&15), K = kt*32 + (l>>4)*8 + (e&7) + (e>>3)*16, e=0..15
__global__ void pack_b_kernel(const float* __restrict__ W, bf16* __restrict__ P,
                              int K, int N) {
    int idx = blockIdx.x * blockDim.x + threadIdx.x;
    if (idx >= K * N) return;
    int frag   = idx >> 9;
    int within = idx & 511;
    int lane   = within >> 4;
    int e      = within & 15;
    int nkt = K >> 5;
    int nt  = frag / nkt;
    int kt  = frag - nt * nkt;
    int n = (nt << 4) + (lane & 15);
    int k = (kt << 5) + ((lane >> 4) << 3) + (e & 7) + ((e >> 3) << 4);
    P[idx] = (bf16)W[(size_t)k * N + n];
}

__global__ void cvt_x_kernel(const float* __restrict__ x, bf16* __restrict__ xb) {
    size_t idx = (size_t)blockIdx.x * blockDim.x + threadIdx.x;
    xb[idx] = (bf16)x[idx];
}

__global__ void init_state_kernel(const float* __restrict__ h0f, const float* __restrict__ c0f,
                                  const float* __restrict__ h0b, const float* __restrict__ c0b,
                                  bf16* __restrict__ hf, bf16* __restrict__ hb,
                                  float* __restrict__ cf, float* __restrict__ cb) {
    int idx = blockIdx.x * blockDim.x + threadIdx.x;
    if (idx >= B_ * H_) return;
    hf[idx] = (bf16)h0f[idx];
    hb[idx] = (bf16)h0b[idx];
    cf[idx] = c0f[idx];
    cb[idx] = c0b[idx];
}

// ---- Input projection: xz[t,b,:] = x[t,b,:] @ W_ih + (b_i + b_h) ----
// grid (512, 64, 2): 64x64 f32 tile per workgroup, 8 waves x 2 WMMA tiles
__global__ __launch_bounds__(256) void gemm_xz_kernel(
    const bf16* __restrict__ Xb,
    const bf16* __restrict__ PWih_f, const bf16* __restrict__ PWih_b,
    const float* __restrict__ bfi, const float* __restrict__ bfh,
    const float* __restrict__ bbi, const float* __restrict__ bbh,
    float* __restrict__ xzf, float* __restrict__ xzb)
{
    const int dir = blockIdx.z;
    const bf16*  PB  = dir ? PWih_b : PWih_f;
    const float* bi  = dir ? bbi : bfi;
    const float* bh  = dir ? bbh : bfh;
    float*       out = dir ? xzb : xzf;

    const int wave = threadIdx.x >> 5;
    const int lane = threadIdx.x & 31;
    const int Mbase = blockIdx.x << 6;
    const int Nbase = blockIdx.y << 6;
    const int m0 = Mbase + ((wave & 3) << 4);
    const int n0 = Nbase + ((wave >> 2) << 5);
    const int ntile0 = n0 >> 4;
    const int nkt = I_ >> 5;

    f32x8 acc0 = {}; f32x8 acc1 = {};
    for (int kt = 0; kt < 32; ++kt) {
        // prefetch the B stream 8 fragments (8 KB) ahead; it is contiguous in kt
        __builtin_prefetch(PB + ((size_t)(ntile0 * nkt + kt + 8) << 9) + (lane << 4), 0, 0);
        __builtin_prefetch(PB + ((size_t)((ntile0 + 1) * nkt + kt + 8) << 9) + (lane << 4), 0, 0);
        bf16x16 a  = load_a_frag(Xb, I_, m0, kt << 5, lane);
        bf16x16 b0 = load_b_frag(PB, ntile0,     kt, nkt, lane);
        bf16x16 b1 = load_b_frag(PB, ntile0 + 1, kt, nkt, lane);
        acc0 = __builtin_amdgcn_wmma_f32_16x16x32_bf16(false, a, false, b0, (short)0, acc0, false, false);
        acc1 = __builtin_amdgcn_wmma_f32_16x16x32_bf16(false, a, false, b1, (short)0, acc1, false, false);
    }
    // C layout: lanes 0-15 -> M = rbase+r, lanes 16-31 -> M = rbase+8+r; N = lane&15
    const int colA = n0 + (lane & 15);
    const int colB = colA + 16;
    const float biasA = bi[colA] + bh[colA];
    const float biasB = bi[colB] + bh[colB];
    const int rbase = m0 + ((lane >> 4) << 3);
    for (int r = 0; r < 8; ++r) {
        size_t row = (size_t)(rbase + r);
        out[row * G_ + colA] = acc0[r] + biasA;
        out[row * G_ + colB] = acc1[r] + biasB;
    }
}

// ---- One recurrent step, both directions. grid (64, 2), block 256 ----
// Block owns h columns [cg*16, cg*16+16) and computes all 4 gates for them.
// wave w: gate g = w>>1, batch-half = w&1 (rows 0..31 or 32..63), 2 WMMA tiles.
// xz_t and c tiles are async-copied into LDS at entry, overlapping the WMMA loop.
__global__ __launch_bounds__(256) void lstm_step_kernel(
    const float* __restrict__ xzf, const float* __restrict__ xzb,
    const bf16* __restrict__ PWhh_f, const bf16* __restrict__ PWhh_b,
    bf16* __restrict__ hfbuf, bf16* __restrict__ hbbuf,
    float* __restrict__ cfbuf, float* __restrict__ cbbuf,
    float* __restrict__ out, int s)
{
    __shared__ float xz_lds[4 * 64 * 16];  // [gate][batch][col] = 16 KB
    __shared__ float zsh[4 * 64 * 16];     // 16 KB
    __shared__ float c_lds[64 * 16];       // 4 KB

    const int dir = blockIdx.y;
    const int cg  = blockIdx.x;                       // h-column group
    const int t   = dir ? (T_ - 1 - s) : s;
    const float* xz  = dir ? xzb : xzf;
    const bf16*  PB  = dir ? PWhh_b : PWhh_f;
    bf16*  hbuf = dir ? hbbuf : hfbuf;
    float* cbuf = dir ? cbbuf : cfbuf;

    const bf16* hprev = hbuf + (size_t)(s & 1) * (B_ * H_);
    bf16*       hnext = hbuf + (size_t)((s + 1) & 1) * (B_ * H_);

    const size_t xzbase = (size_t)t * (B_ * G_);

    // ---- issue async global->LDS prefetch of xz tile (16 KB) + c tile (4 KB) ----
    {
        const int tid = threadIdx.x;
        for (int j = 0; j < 4; ++j) {              // 1024 chunks of 16 B, 4/thread
            int f   = (tid + (j << 8)) << 2;       // flat float index
            int gg  = f >> 10;                     // gate
            int row = (f >> 4) & 63;               // batch
            int c4  = f & 15;                      // col (multiple of 4)
            copy16_g2l(&xz_lds[f],
                       xz + xzbase + (size_t)row * G_ + (gg << 10) + (cg << 4) + c4);
        }
        {
            int f   = tid << 2;                    // 256 chunks, 1/thread
            int row = f >> 4;
            int c4  = f & 15;
            copy16_g2l(&c_lds[f], cbuf + (size_t)row * H_ + (cg << 4) + c4);
        }
    }

    const int wave = threadIdx.x >> 5;
    const int lane = threadIdx.x & 31;
    const int g     = wave >> 1;
    const int m0    = (wave & 1) << 5;                // batch rows 0 or 32
    const int ntile = (g << 6) + cg;                  // N-tile in W_hh (N = 4096)
    const int nkt   = H_ >> 5;

    f32x8 acc0 = {}; f32x8 acc1 = {};
    for (int kt = 0; kt < 32; ++kt) {
        // B stream is contiguous in kt; pull it into L0 8 fragments ahead
        __builtin_prefetch(PB + ((size_t)(ntile * nkt + kt + 8) << 9) + (lane << 4), 0, 0);
        bf16x16 a0 = load_a_frag(hprev, H_, m0,      kt << 5, lane);
        bf16x16 a1 = load_a_frag(hprev, H_, m0 + 16, kt << 5, lane);
        bf16x16 b  = load_b_frag(PB, ntile, kt, nkt, lane);
        acc0 = __builtin_amdgcn_wmma_f32_16x16x32_bf16(false, a0, false, b, (short)0, acc0, false, false);
        acc1 = __builtin_amdgcn_wmma_f32_16x16x32_bf16(false, a1, false, b, (short)0, acc1, false, false);
    }

    // prefetched tiles are now (almost certainly) resident; wait + publish
    wait_g2l();
    __syncthreads();

    // z = acc + xz_t, exchange through LDS so each thread sees all 4 gates
    const int col = lane & 15;
    const int r0  = m0 + ((lane >> 4) << 3);
    for (int r = 0; r < 8; ++r) {
        int rowA = r0 + r;
        int rowB = rowA + 16;
        int ia = (g << 10) + (rowA << 4) + col;
        int ib = (g << 10) + (rowB << 4) + col;
        zsh[ia] = acc0[r] + xz_lds[ia];
        zsh[ib] = acc1[r] + xz_lds[ib];
    }
    __syncthreads();

    // gates + state update: 64x16 elements, 4 per thread
    for (int j = 0; j < 4; ++j) {
        int idx = threadIdx.x + (j << 8);
        int b = idx >> 4;
        int c = idx & 15;
        float zi = zsh[(0 << 10) + idx];
        float zf = zsh[(1 << 10) + idx];
        float zg = zsh[(2 << 10) + idx];
        float zo = zsh[(3 << 10) + idx];
        float ig = 1.0f / (1.0f + __expf(-zi));
        float fg = 1.0f / (1.0f + __expf(-zf));
        float gg = 2.0f / (1.0f + __expf(-2.0f * zg)) - 1.0f;  // tanh
        float og = 1.0f / (1.0f + __expf(-zo));
        int hcol = (cg << 4) + c;
        size_t cidx = (size_t)b * H_ + hcol;
        float cn = fg * c_lds[idx] + ig * gg;
        float hn = og * cn;                            // reference: h = o * c (no tanh)
        cbuf[cidx]  = cn;
        hnext[cidx] = (bf16)hn;
        out[(size_t)t * (B_ * 2 * H_) + (size_t)b * (2 * H_) + (size_t)dir * H_ + hcol] = hn;
    }
}

extern "C" void kernel_launch(void* const* d_in, const int* in_sizes, int n_in,
                              void* d_out, int out_size, void* d_ws, size_t ws_size,
                              hipStream_t stream) {
    (void)in_sizes; (void)n_in; (void)out_size; (void)ws_size;
    const float* x     = (const float*)d_in[0];
    const float* Wf_ih = (const float*)d_in[1];
    const float* Wf_hh = (const float*)d_in[2];
    const float* bf_i  = (const float*)d_in[3];
    const float* bf_h  = (const float*)d_in[4];
    const float* Wb_ih = (const float*)d_in[5];
    const float* Wb_hh = (const float*)d_in[6];
    const float* bb_i  = (const float*)d_in[7];
    const float* bb_h  = (const float*)d_in[8];
    const float* h0f   = (const float*)d_in[9];
    const float* c0f   = (const float*)d_in[10];
    const float* h0b   = (const float*)d_in[11];
    const float* c0b   = (const float*)d_in[12];
    float* out = (float*)d_out;

    // workspace carve-up (256 B aligned)
    char* ws = (char*)d_ws;
    size_t off = 0;
    auto carve = [&](size_t bytes) { void* p = ws + off; off = (off + bytes + 255) & ~(size_t)255; return p; };
    const size_t SZ_XZ = (size_t)T_ * B_ * G_ * sizeof(float);   // 512 MB
    float* xzf    = (float*)carve(SZ_XZ);
    float* xzb    = (float*)carve(SZ_XZ);
    bf16* pwf_ih  = (bf16*)carve((size_t)I_ * G_ * sizeof(bf16));
    bf16* pwb_ih  = (bf16*)carve((size_t)I_ * G_ * sizeof(bf16));
    bf16* pwf_hh  = (bf16*)carve((size_t)H_ * G_ * sizeof(bf16));
    bf16* pwb_hh  = (bf16*)carve((size_t)H_ * G_ * sizeof(bf16));
    bf16* xbf     = (bf16*)carve((size_t)T_ * B_ * I_ * sizeof(bf16));
    bf16* hfbuf   = (bf16*)carve((size_t)2 * B_ * H_ * sizeof(bf16));  // double buffer
    bf16* hbbuf   = (bf16*)carve((size_t)2 * B_ * H_ * sizeof(bf16));
    float* cfbuf  = (float*)carve((size_t)B_ * H_ * sizeof(float));
    float* cbbuf  = (float*)carve((size_t)B_ * H_ * sizeof(float));

    // 1) pack weights into WMMA B-fragment order (bf16)
    {
        int blocks = (I_ * G_) / 256;
        pack_b_kernel<<<blocks, 256, 0, stream>>>(Wf_ih, pwf_ih, I_, G_);
        pack_b_kernel<<<blocks, 256, 0, stream>>>(Wb_ih, pwb_ih, I_, G_);
        pack_b_kernel<<<blocks, 256, 0, stream>>>(Wf_hh, pwf_hh, H_, G_);
        pack_b_kernel<<<blocks, 256, 0, stream>>>(Wb_hh, pwb_hh, H_, G_);
    }
    // 2) x -> bf16
    cvt_x_kernel<<<(T_ * B_ * I_) / 256, 256, 0, stream>>>(x, xbf);
    // 3) initial h (bf16, slot 0) and c (f32)
    init_state_kernel<<<(B_ * H_) / 256, 256, 0, stream>>>(h0f, c0f, h0b, c0b,
                                                           hfbuf, hbbuf, cfbuf, cbbuf);
    // 4) time-parallel input projection for both directions
    gemm_xz_kernel<<<dim3((T_ * B_) / 64, G_ / 64, 2), 256, 0, stream>>>(
        xbf, pwf_ih, pwb_ih, bf_i, bf_h, bb_i, bb_h, xzf, xzb);
    // 5) sequential recurrence, both directions per launch
    for (int s = 0; s < T_; ++s) {
        lstm_step_kernel<<<dim3(H_ / 16, 2), 256, 0, stream>>>(
            xzf, xzb, pwf_hh, pwb_hh, hfbuf, hbbuf, cfbuf, cbbuf, out, s);
    }
}
